// MLA_72009421685026
// MI455X (gfx1250) — matrix operand order repaired and checked by reading, not compiled, not run on hardware
//
#include <hip/hip_runtime.h>
#include <hip/hip_bf16.h>
#include <math.h>

// ---------------- MLA dims ----------------
#define T_TOK 1024
#define D_MODEL 7168
#define N_HEADS 128
#define A_Q 1536
#define A_KV 512
#define QK_NOPE 128
#define QK_ROPE 64
#define V_DIM 128
#define QK_HEAD 192                 // 128 + 64
#define KV_DOWN (A_KV + QK_ROPE)    // 576
#define KV_DOWN_PAD 640             // padded to a multiple of 128
// combined scale: QK_HEAD^-0.5 * PAD_H^-0.5 = 192^-0.5 / 16
#define Q_SCALE 0.00451054899f

typedef __bf16 bf16;
typedef __bf16 bf16x8  __attribute__((ext_vector_type(8)));
typedef __bf16 bf16x16 __attribute__((ext_vector_type(16)));
typedef float  v8f     __attribute__((ext_vector_type(8)));

#if defined(__gfx1250__) && __has_builtin(__builtin_amdgcn_tensor_load_to_lds)
#define USE_TDM 1
#else
#define USE_TDM 0
#endif

__device__ __forceinline__ bf16x16 cat16(bf16x8 lo, bf16x8 hi) {
  return __builtin_shufflevector(lo, hi, 0,1,2,3,4,5,6,7,8,9,10,11,12,13,14,15);
}
__device__ __forceinline__ v8f wmma_bf16f32(bf16x16 a, bf16x16 b, v8f c) {
  return __builtin_amdgcn_wmma_f32_16x16x32_bf16(false, a, false, b, (short)0, c,
                                                 false, false);
}

#if USE_TDM
typedef unsigned int uint4v __attribute__((ext_vector_type(4)));
typedef int int8v __attribute__((ext_vector_type(8)));
typedef int int4v __attribute__((ext_vector_type(4)));

// Issue a TDM load of a 2-D bf16 tile (tile_k contiguous elems x tile_m rows,
// row stride = stride_elems) from global memory into LDS at lds_addr.
// D# bitfields per CDNA5 ISA.
// This toolchain exposes the 6-arg builtin:
//   (uint32x4 g0, int32x8 g1, int32x4 g2, int32x4 g3, int32x8 pad, i32 cpol)
__device__ __forceinline__ void tdm_load_tile_2d(unsigned lds_addr,
                                                 const void* gptr, int tile_k,
                                                 int tile_m, long stride_elems) {
  unsigned long long ga = (unsigned long long)(uintptr_t)gptr;
  uint4v g0;
  g0.x = 1u;                                   // count=1 (valid user D#)
  g0.y = lds_addr;                             // lds_addr (bytes)
  g0.z = (unsigned)(ga & 0xffffffffu);         // global_addr[31:0]
  g0.w = (unsigned)((ga >> 32) & 0x01ffffffu)  // global_addr[56:32]
         | (2u << 30);                         // type = 2 ("image")
  int8v g1;
  g1[0] = 0x00010000;                                   // data_size=1 (2B)
  g1[1] = (tile_k & 0xffff) << 16;                      // tensor_dim0[15:0]
  g1[2] = ((unsigned)tile_k >> 16) | ((tile_m & 0xffff) << 16);  // dim0hi|dim1lo
  g1[3] = ((unsigned)tile_m >> 16) | ((tile_k & 0xffff) << 16);  // dim1hi|tile_dim0
  g1[4] = (tile_m & 0xffff);                            // tile_dim1 | tile_dim2=0
  g1[5] = (int)(stride_elems & 0xffffffffL);            // dim0_stride[31:0]
  g1[6] = (int)((stride_elems >> 32) & 0xffffL);        // dim0_stride[47:32]
  g1[7] = 0;
  int4v z4 = {0, 0, 0, 0};                              // groups 2/3 unused (2D)
  int8v z8 = {0, 0, 0, 0, 0, 0, 0, 0};
  __builtin_amdgcn_tensor_load_to_lds(g0, g1, z4, z4, z8, 0);
}
#endif

// ---------------- yarn rope inverse frequency (computed analytically) ------
__device__ __forceinline__ float yarn_inv_freq(int i) {   // i in [0, 32)
  const float dim = 64.f, theta = 10000.f, factor = 40.f;
  const float two_pi = 6.283185307f;
  float pf        = __powf(theta, (2.f * (float)i) / dim);
  float inv_extra = 1.f / pf;
  float inv_inter = 1.f / (factor * pf);
  float lt   = logf(theta);
  float low  = fmaxf(floorf(dim * logf(4096.f / (32.f * two_pi)) / (2.f * lt)), 0.f);
  float high = fminf(ceilf (dim * logf(4096.f / ( 1.f * two_pi)) / (2.f * lt)), 63.f);
  float ramp = ((float)i - low) / fmaxf(high - low, 0.001f);
  ramp = fminf(fmaxf(ramp, 0.f), 1.f);
  return inv_inter * ramp + inv_extra * (1.f - ramp);
}

// ---------------- f32 -> bf16 conversion ----------------
__global__ void cvt_f32_to_bf16(const float* __restrict__ s, bf16* __restrict__ d,
                                size_t n) {
  size_t i = (size_t)blockIdx.x * blockDim.x + threadIdx.x;
  size_t stride = (size_t)gridDim.x * blockDim.x;
  for (; i < n; i += stride) d[i] = (bf16)s[i];
}

// f32 -> bf16 with row padding (zero fill) so GEMM B tiles never need guards
__global__ void cvt_f32_to_bf16_pad(const float* __restrict__ s,
                                    bf16* __restrict__ d, int rows, int cols,
                                    int ldd) {
  size_t total = (size_t)rows * ldd;
  size_t stride = (size_t)gridDim.x * blockDim.x;
  for (size_t i = (size_t)blockIdx.x * blockDim.x + threadIdx.x; i < total;
       i += stride) {
    int r = (int)(i / ldd), c = (int)(i % ldd);
    d[i] = (c < cols) ? (bf16)s[(size_t)r * cols + c] : (bf16)0.f;
  }
}

// ---------------- generic bf16 GEMM: C(f32) = A(bf16) x B(bf16) ------------
// block tile 128x128, 8 waves -> each wave 32(M) x 64(N) = 2x4 WMMA tiles.
// Caller guarantees every B row has >= ceil(N/128)*128 readable columns.
__launch_bounds__(256, 1)
__global__ void gemm_bf16(const bf16* __restrict__ A, const bf16* __restrict__ B,
                          float* __restrict__ C, int N, int K,
                          int lda, int ldb, int ldc) {
  __shared__ __align__(16) bf16 sA[128 * 32];   // [row][k]
  __shared__ __align__(16) bf16 sBt[128 * 32];  // [n][k]  (transposed B)
  const int n0 = blockIdx.x * 128, m0 = blockIdx.y * 128;
  const int t = threadIdx.x, lane = t & 31, wave = t >> 5;
  const int wm = wave >> 1, wn = wave & 1;
  const int half = lane >> 4, lc = lane & 15;

  v8f acc[2][4];
#pragma unroll
  for (int i = 0; i < 2; ++i)
#pragma unroll
    for (int j = 0; j < 4; ++j) acc[i][j] = (v8f){0, 0, 0, 0, 0, 0, 0, 0};

  for (int k0 = 0; k0 < K; k0 += 32) {
#if USE_TDM
    // A tile via Tensor Data Mover: one wave issues the DMA (EXEC ignored),
    // everyone waits on TENSORcnt before the block barrier.
    if (wave == 0) {
      tdm_load_tile_2d((unsigned)(uintptr_t)&sA[0],
                       &A[(size_t)m0 * lda + k0], 32, 128, (long)lda);
    }
#else
    // manual A tile stage (128x32), 8 halves per chunk, 2 chunks per thread
#pragma unroll
    for (int i = 0; i < 2; ++i) {
      int c = t * 2 + i, r = c >> 2, s = (c & 3) * 8;
      *(bf16x8*)&sA[r * 32 + s] =
          *(const bf16x8*)&A[(size_t)(m0 + r) * lda + k0 + s];
    }
#endif
    // stage B tile transposed (read coalesced along n, scatter along k)
#pragma unroll
    for (int i = 0; i < 2; ++i) {
      int c = t * 2 + i, kr = c >> 4, ns = (c & 15) * 8;
      bf16x8 vv = *(const bf16x8*)&B[(size_t)(k0 + kr) * ldb + n0 + ns];
#pragma unroll
      for (int h = 0; h < 8; ++h) sBt[(ns + h) * 32 + kr] = vv[h];
    }
#if USE_TDM
    asm volatile("s_wait_tensorcnt 0" ::: "memory");
#endif
    __syncthreads();
    if (k0 + 32 < K) {  // hint next K slab into L2 (global_prefetch_b8)
      __builtin_prefetch(&A[(size_t)(m0 + (t & 127)) * lda + k0 + 32], 0, 0);
      __builtin_prefetch(&B[(size_t)(k0 + 32 + (t & 31)) * ldb + n0 + (t >> 5) * 16], 0, 0);
    }
    // fragments (ISA 16-bit A / B per-lane layouts)
    bf16x16 af[2], bfg[4];
#pragma unroll
    for (int i = 0; i < 2; ++i) {
      const bf16* p = &sA[(wm * 32 + i * 16 + lc) * 32 + half * 8];
      af[i] = cat16(*(const bf16x8*)p, *(const bf16x8*)(p + 16));
    }
#pragma unroll
    for (int j = 0; j < 4; ++j) {
      const bf16* p = &sBt[(wn * 64 + j * 16 + lc) * 32 + half * 8];
      bfg[j] = cat16(*(const bf16x8*)p, *(const bf16x8*)(p + 16));
    }
#pragma unroll
    for (int i = 0; i < 2; ++i)
#pragma unroll
      for (int j = 0; j < 4; ++j)
        acc[i][j] = wmma_bf16f32(af[i], bfg[j], acc[i][j]);
    __syncthreads();
  }
  // store C (f32, C-layout: VGPR v -> row v + 8*half)
#pragma unroll
  for (int i = 0; i < 2; ++i) {
    const int rb = m0 + wm * 32 + i * 16 + half * 8;
#pragma unroll
    for (int j = 0; j < 4; ++j) {
      const int col = n0 + wn * 64 + j * 16 + lc;
      if (col < N) {
#pragma unroll
        for (int v = 0; v < 8; ++v)
          C[(size_t)(rb + v) * ldc + col] = acc[i][j][v];
      }
    }
  }
}

// ---------------- RMS norm over rows ----------------
__global__ void rmsnorm_rows(const float* __restrict__ in,
                             const float* __restrict__ scale,
                             bf16* __restrict__ out, int cols, int ld_in) {
  const int row = blockIdx.x;
  const float* r = in + (size_t)row * ld_in;
  float ss = 0.f;
  for (int i = threadIdx.x; i < cols; i += 256) { float v = r[i]; ss += v * v; }
#pragma unroll
  for (int off = 16; off; off >>= 1) ss += __shfl_xor(ss, off, 32);
  __shared__ float part[8];
  if ((threadIdx.x & 31) == 0) part[threadIdx.x >> 5] = ss;
  __syncthreads();
  float tot = 0.f;
#pragma unroll
  for (int w = 0; w < 8; ++w) tot += part[w];
  const float inv = rsqrtf(tot / (float)cols + 1e-6f);
  for (int i = threadIdx.x; i < cols; i += 256)
    out[(size_t)row * cols + i] = (bf16)(r[i] * inv * scale[i]);
}

// ---------------- q epilogue: rope + scale, layout [head][t][192] ----------
__global__ void assemble_q(const float* __restrict__ qup, const int* __restrict__ pos,
                           bf16* __restrict__ Qh) {
  const int t = blockIdx.x, n = blockIdx.y, h = threadIdx.x;  // blockDim 192
  const float* src = qup + (size_t)t * (N_HEADS * QK_HEAD) + n * QK_HEAD;
  float v;
  if (h < QK_NOPE) {
    v = src[h];
  } else {
    const int i = h - QK_NOPE;
    const float p = (float)pos[t];
    if (i < 32) {
      float ang = p * yarn_inv_freq(i);
      v = src[QK_NOPE + i] * __cosf(ang) - src[QK_NOPE + 32 + i] * __sinf(ang);
    } else {
      int ii = i - 32;
      float ang = p * yarn_inv_freq(ii);
      v = src[QK_NOPE + i] * __cosf(ang) + src[QK_NOPE + ii] * __sinf(ang);
    }
  }
  Qh[((size_t)n * T_TOK + t) * QK_HEAD + h] = (bf16)(v * Q_SCALE);
}

// ---------------- kv epilogue: K [head][s][192], V transposed [head][d][s] -
__global__ void assemble_kv(const float* __restrict__ kvup,
                            const float* __restrict__ kva,
                            const int* __restrict__ pos,
                            bf16* __restrict__ Kh, bf16* __restrict__ Vt) {
  const int s = blockIdx.x, n = blockIdx.y, h = threadIdx.x;  // blockDim 256
  const float* src = kvup + (size_t)s * (N_HEADS * 256) + n * 256;
  if (h < QK_NOPE) {
    Kh[((size_t)n * T_TOK + s) * QK_HEAD + h] = (bf16)src[h];
  } else {
    const int d = h - QK_NOPE;
    Vt[((size_t)n * V_DIM + d) * T_TOK + s] = (bf16)src[h];
  }
  if (h < QK_ROPE) {  // shared k-rope broadcast across heads (no q scale here)
    const float* kr = kva + (size_t)s * KV_DOWN + A_KV;
    const float p = (float)pos[s];
    float v;
    if (h < 32) {
      float ang = p * yarn_inv_freq(h);
      v = kr[h] * __cosf(ang) - kr[32 + h] * __sinf(ang);
    } else {
      int ii = h - 32;
      float ang = p * yarn_inv_freq(ii);
      v = kr[h] * __cosf(ang) + kr[ii] * __sinf(ang);
    }
    Kh[((size_t)n * T_TOK + s) * QK_HEAD + QK_NOPE + h] = (bf16)v;
  }
}

// ---------------- flash attention: 16 queries per wave, causal -------------
__launch_bounds__(256, 1)
__global__ void attn_kernel(const bf16* __restrict__ Qh, const bf16* __restrict__ Kh,
                            const bf16* __restrict__ Vt, bf16* __restrict__ Oa) {
  __shared__ __align__(16) bf16 sP[8][16 * 32];  // per-wave P staging
  const int n = blockIdx.y, qb = blockIdx.x;
  const int t = threadIdx.x, wave = t >> 5, lane = t & 31;
  const int half = lane >> 4, lc = lane & 15;
  const int q0 = qb * 128 + wave * 16;
  const bf16* Qp = Qh + (size_t)n * T_TOK * QK_HEAD;
  const bf16* Kp = Kh + (size_t)n * T_TOK * QK_HEAD;
  const bf16* Vp = Vt + (size_t)n * V_DIM * T_TOK;

  // preload 6 Q A-fragments (covers h = 0..191)
  bf16x16 aq[6];
  {
    const bf16* qr = Qp + (size_t)(q0 + lc) * QK_HEAD;
#pragma unroll
    for (int h = 0; h < 6; ++h) {
      const bf16* b = qr + h * 32 + half * 8;
      aq[h] = cat16(*(const bf16x8*)b, *(const bf16x8*)(b + 16));
    }
  }
  v8f o[8];
  float mrow[8], lrow[8];
#pragma unroll
  for (int j = 0; j < 8; ++j) {
    o[j] = (v8f){0, 0, 0, 0, 0, 0, 0, 0};
    mrow[j] = -1e30f;
    lrow[j] = 0.f;
  }

  for (int kb = 0; kb <= q0 + 15; kb += 32) {   // wave-uniform causal bound
    v8f s0 = (v8f){0, 0, 0, 0, 0, 0, 0, 0};
    v8f s1 = (v8f){0, 0, 0, 0, 0, 0, 0, 0};
#pragma unroll
    for (int h = 0; h < 6; ++h) {
      const bf16* b0 = Kp + (size_t)(kb + lc) * QK_HEAD + h * 32 + half * 8;
      s0 = wmma_bf16f32(aq[h], cat16(*(const bf16x8*)b0, *(const bf16x8*)(b0 + 16)), s0);
      const bf16* b1 = Kp + (size_t)(kb + 16 + lc) * QK_HEAD + h * 32 + half * 8;
      s1 = wmma_bf16f32(aq[h], cat16(*(const bf16x8*)b1, *(const bf16x8*)(b1 + 16)), s1);
    }
    // online softmax per output row (row = q0 + v + 8*half; one row lives in
    // one VGPR across a 16-lane half => xor-masks 1..8 reduce it)
    bf16* pw = &sP[wave][0];
#pragma unroll
    for (int v = 0; v < 8; ++v) {
      const int row = q0 + v + 8 * half;
      float x0 = (kb + lc > row) ? -1e30f : s0[v];
      float x1 = (kb + 16 + lc > row) ? -1e30f : s1[v];
      float rm = fmaxf(x0, x1);
#pragma unroll
      for (int off = 8; off; off >>= 1) rm = fmaxf(rm, __shfl_xor(rm, off, 32));
      const float mn = fmaxf(mrow[v], rm);
      const float alpha = __expf(mrow[v] - mn);
      const float p0 = __expf(x0 - mn), p1 = __expf(x1 - mn);
      float rs = p0 + p1;
#pragma unroll
      for (int off = 8; off; off >>= 1) rs += __shfl_xor(rs, off, 32);
      lrow[v] = lrow[v] * alpha + rs;
      mrow[v] = mn;
#pragma unroll
      for (int j = 0; j < 8; ++j) o[j][v] = o[j][v] * alpha;
      const int r32 = (v + 8 * half) * 32;
      pw[r32 + lc] = (bf16)p0;
      pw[r32 + 16 + lc] = (bf16)p1;
    }
    asm volatile("s_wait_dscnt 0" ::: "memory");  // P store -> P load (same wave)
    const bf16* pr = pw + lc * 32 + half * 8;
    bf16x16 ap = cat16(*(const bf16x8*)pr, *(const bf16x8*)(pr + 16));
    // P x V : V pre-transposed [d][s] so B-fragments are contiguous loads
#pragma unroll
    for (int j = 0; j < 8; ++j) {
      const bf16* vb = Vp + (size_t)(j * 16 + lc) * T_TOK + kb + half * 8;
      o[j] = wmma_bf16f32(ap, cat16(*(const bf16x8*)vb, *(const bf16x8*)(vb + 16)), o[j]);
    }
    asm volatile("" ::: "memory");
  }
  // normalize + store to [t][head*128 + d] (A-matrix for o-proj GEMM)
#pragma unroll
  for (int j = 0; j < 8; ++j) {
#pragma unroll
    for (int v = 0; v < 8; ++v) {
      const int row = q0 + v + 8 * half;
      Oa[(size_t)row * (N_HEADS * V_DIM) + n * V_DIM + j * 16 + lc] =
          (bf16)(o[j][v] / lrow[v]);
    }
  }
}

// ---------------- host orchestration ----------------
extern "C" void kernel_launch(void* const* d_in, const int* in_sizes, int n_in,
                              void* d_out, int out_size, void* d_ws, size_t ws_size,
                              hipStream_t stream) {
  const float* x    = (const float*)d_in[0];
  const int*   pos  = (const int*)d_in[1];
  const float* wqd  = (const float*)d_in[2];
  const float* qns  = (const float*)d_in[3];
  const float* wqu  = (const float*)d_in[4];
  const float* wkvd = (const float*)d_in[5];
  const float* kvns = (const float*)d_in[6];
  const float* wkvu = (const float*)d_in[7];
  const float* wo   = (const float*)d_in[8];
  float* out = (float*)d_out;

  char* ws = (char*)d_ws;
  size_t off = 0;
  auto alloc = [&](size_t bytes) -> void* {
    void* p = ws + off;
    off += (bytes + 255) & ~(size_t)255;
    return p;
  };
  bf16* xb     = (bf16*)alloc((size_t)T_TOK * D_MODEL * 2);
  bf16* wqd_b  = (bf16*)alloc((size_t)D_MODEL * A_Q * 2);
  bf16* wqu_b  = (bf16*)alloc((size_t)A_Q * N_HEADS * QK_HEAD * 2);
  bf16* wkvd_b = (bf16*)alloc((size_t)D_MODEL * KV_DOWN_PAD * 2);  // padded
  bf16* wkvu_b = (bf16*)alloc((size_t)A_KV * N_HEADS * 256 * 2);
  bf16* wo_b   = (bf16*)alloc((size_t)N_HEADS * V_DIM * D_MODEL * 2);
  float* qa    = (float*)alloc((size_t)T_TOK * A_Q * 4);
  float* kva   = (float*)alloc((size_t)T_TOK * KV_DOWN * 4);
  bf16* qn     = (bf16*)alloc((size_t)T_TOK * A_Q * 2);
  bf16* cn     = (bf16*)alloc((size_t)T_TOK * A_KV * 2);
  float* qup   = (float*)alloc((size_t)T_TOK * N_HEADS * QK_HEAD * 4);
  float* kvup  = (float*)alloc((size_t)T_TOK * N_HEADS * 256 * 4);
  bf16* Qh     = (bf16*)alloc((size_t)N_HEADS * T_TOK * QK_HEAD * 2);
  bf16* Kh     = (bf16*)alloc((size_t)N_HEADS * T_TOK * QK_HEAD * 2);
  bf16* Vt     = (bf16*)alloc((size_t)N_HEADS * V_DIM * T_TOK * 2);
  bf16* Oa     = (bf16*)alloc((size_t)T_TOK * N_HEADS * V_DIM * 2);

  auto cvt = [&](const float* s, bf16* d, size_t n) {
    size_t blocks = (n + 255) / 256;
    int grid = blocks > 2048 ? 2048 : (int)blocks;
    cvt_f32_to_bf16<<<grid, 256, 0, stream>>>(s, d, n);
  };
  cvt(x,    xb,     (size_t)T_TOK * D_MODEL);
  cvt(wqd,  wqd_b,  (size_t)D_MODEL * A_Q);
  cvt(wqu,  wqu_b,  (size_t)A_Q * N_HEADS * QK_HEAD);
  cvt(wkvu, wkvu_b, (size_t)A_KV * N_HEADS * 256);
  cvt(wo,   wo_b,   (size_t)N_HEADS * V_DIM * D_MODEL);
  // kv_down weight converted with zero padding to 640 columns
  cvt_f32_to_bf16_pad<<<2048, 256, 0, stream>>>(wkvd, wkvd_b, D_MODEL, KV_DOWN,
                                                KV_DOWN_PAD);

  // q_down: [1024,7168] x [7168,1536]
  gemm_bf16<<<dim3(A_Q / 128, T_TOK / 128), 256, 0, stream>>>(
      xb, wqd_b, qa, A_Q, D_MODEL, D_MODEL, A_Q, A_Q);
  // kv_down: [1024,7168] x [7168,576] (B padded to 640)
  gemm_bf16<<<dim3((KV_DOWN + 127) / 128, T_TOK / 128), 256, 0, stream>>>(
      xb, wkvd_b, kva, KV_DOWN, D_MODEL, D_MODEL, KV_DOWN_PAD, KV_DOWN);
  // RMS norms
  rmsnorm_rows<<<T_TOK, 256, 0, stream>>>(qa, qns, qn, A_Q, A_Q);
  rmsnorm_rows<<<T_TOK, 256, 0, stream>>>(kva, kvns, cn, A_KV, KV_DOWN);
  // q_up: [1024,1536] x [1536,24576]
  gemm_bf16<<<dim3((N_HEADS * QK_HEAD) / 128, T_TOK / 128), 256, 0, stream>>>(
      qn, wqu_b, qup, N_HEADS * QK_HEAD, A_Q, A_Q, N_HEADS * QK_HEAD,
      N_HEADS * QK_HEAD);
  // kv_up: [1024,512] x [512,32768]
  gemm_bf16<<<dim3((N_HEADS * 256) / 128, T_TOK / 128), 256, 0, stream>>>(
      cn, wkvu_b, kvup, N_HEADS * 256, A_KV, A_KV, N_HEADS * 256, N_HEADS * 256);
  // epilogues: rope + relayouts
  assemble_q<<<dim3(T_TOK, N_HEADS), 192, 0, stream>>>(qup, pos, Qh);
  assemble_kv<<<dim3(T_TOK, N_HEADS), 256, 0, stream>>>(kvup, kva, pos, Kh, Vt);
  // flash attention
  attn_kernel<<<dim3(T_TOK / 128, N_HEADS), 256, 0, stream>>>(Qh, Kh, Vt, Oa);
  // o_proj: [1024,16384] x [16384,7168] -> d_out (f32)
  gemm_bf16<<<dim3(D_MODEL / 128, T_TOK / 128), 256, 0, stream>>>(
      Oa, wo_b, out, D_MODEL, N_HEADS * V_DIM, N_HEADS * V_DIM, D_MODEL, D_MODEL);
}
